// Kernel_13877190235998553350_60842506715274
// MI455X (gfx1250) — matrix operand order repaired and checked by reading, not compiled, not run on hardware
//
#include <hip/hip_runtime.h>

// ---------------------------------------------------------------------------
// CDNA5 / gfx1250 implementation.
// N=16, C=256, H=W=64, E=C/8=32, F=3C=768, K5=E*W=2048, P=H*W=4096.
// Heavy einsums: LDS-tiled bf16 WMMA GEMMs with K-major B operands.
// Tile staging uses the CDNA5 Tensor Data Mover: one tensor_load_to_lds per
// 128x32 tile (issued by wave 0, completion via s_wait_tensorcnt/TENSORcnt),
// fragments load as ds_load_b128. Double-buffered LDS, 1 barrier per K step.
// ---------------------------------------------------------------------------

typedef __attribute__((ext_vector_type(16))) __bf16        v16bf;
typedef __attribute__((ext_vector_type(8)))  float         v8f;
typedef __attribute__((ext_vector_type(8)))  unsigned int  v8u;
typedef __attribute__((ext_vector_type(4)))  unsigned int  v4u;
typedef __attribute__((ext_vector_type(8)))  int           v8i;
typedef __attribute__((ext_vector_type(4)))  int           v4i;

__device__ __forceinline__ unsigned short f2bf(float f) {
    unsigned int u = __float_as_uint(f);
    u = (u + 0x7FFFu + ((u >> 16) & 1u)) >> 16;   // round-to-nearest-even
    return (unsigned short)u;
}

// Generic (flat) shared pointer -> 32-bit LDS offset (low 32 bits).
__device__ __forceinline__ unsigned int lds_off(const void* p) {
    return (unsigned int)(size_t)p;
}

// ---------------------------------------------------------------------------
// TDM tile load: 128 rows x 32 bf16 (64B) per row, row stride ld_elems,
// appended contiguously into LDS (matches both the row-major A tile and the
// column-packed B tile layouts).  D# per CDNA5 ISA §8.3/§8.4:
//   group0: count=1 | lds_addr | global_addr[56:0] | type=2
//   group1: wg_mask=0, data_size=1 (2B), tensor_dim0=ld, tensor_dim1=128,
//           tile_dim0=32, tile_dim1=128, tensor_dim0_stride=ld
//   groups 2/3 (+ extra group on this toolchain): zero (2-D tensor)
// ---------------------------------------------------------------------------
__device__ __forceinline__ void tdm_load_tile(void* ldsbuf, const void* gptr,
                                              unsigned int ld_elems) {
    const unsigned long long ga = (unsigned long long)(size_t)gptr;
    v4u g0;
    g0[0] = 1u;                                        // count=1 (valid user D#)
    g0[1] = lds_off(ldsbuf);                           // lds_addr (bytes)
    g0[2] = (unsigned int)(ga & 0xffffffffu);          // global_addr[31:0]
    g0[3] = (unsigned int)((ga >> 32) & 0x01ffffffu)   // global_addr[56:32]
            | (2u << 30);                              // type = 2 ("image")
    v8i g1;
    g1[0] = (int)(1u << 16);                           // mask=0, data_size=2B
    g1[1] = (int)((ld_elems & 0xffffu) << 16);         // tensor_dim0[15:0]
    g1[2] = (int)(((ld_elems >> 16) & 0xffffu)         // tensor_dim0[31:16]
            | (128u << 16));                           // tensor_dim1[15:0]=128
    g1[3] = (int)(32u << 16);                          // tile_dim0 = 32
    g1[4] = 128;                                       // tile_dim1=128, dim2=0
    g1[5] = (int)ld_elems;                             // dim0_stride[31:0]
    g1[6] = 0;                                         // stride hi bits
    g1[7] = 0;
    const v4i z4 = {0, 0, 0, 0};
    const v8i z8 = {0, 0, 0, 0, 0, 0, 0, 0};
    __builtin_amdgcn_tensor_load_to_lds(g0, g1, z4, z4, z8, 0);
}

__device__ __forceinline__ void wait_tensor0() {
    __builtin_amdgcn_s_wait_tensorcnt(0);
}

// A fragment: 16x32 bf16, row-major LDS tile (lda = 32).
//   lanes 0-15 : row = lane,      K = 0..15
//   lanes 16-31: row = lane - 16, K = 16..31
__device__ __forceinline__ v16bf load_frag_a(const unsigned short* As, int row,
                                             int lane) {
    const int r  = row + (lane & 15);
    const int kb = (lane >> 4) * 16;
    v4u lo = *(const v4u*)(As + r * 32 + kb);
    v4u hi = *(const v4u*)(As + r * 32 + kb + 8);
    v8u u  = __builtin_shufflevector(lo, hi, 0, 1, 2, 3, 4, 5, 6, 7);
    return __builtin_bit_cast(v16bf, u);
}

// B fragment from column-major packed LDS tile Bt[col][32 K values].
//   lanes 0-15 : col = lane,      K = 0..15
//   lanes 16-31: col = lane - 16, K = 16..31
__device__ __forceinline__ v16bf load_frag_b(const unsigned short* Bt, int col,
                                             int lane) {
    const int cc = col + (lane & 15);
    const int kb = (lane >> 4) * 16;
    v8u u = *(const v8u*)(Bt + cc * 32 + kb);
    return __builtin_bit_cast(v16bf, u);
}

__device__ __forceinline__ v8f wmma_bf16(v16bf a, v16bf b, v8f c) {
    return __builtin_amdgcn_wmma_f32_16x16x32_bf16(
        false, a, false, b, (short)0, c, false, false);
}

// ---------------------------------------------------------------------------
// p5T[f][k] = bf16(p5[k][f])   (K-major B for the t5 GEMM; k = e*64+w)
// ---------------------------------------------------------------------------
__global__ void build_p5T_kernel(const float* __restrict__ p5,
                                 unsigned short* __restrict__ p5T) {
    const int gid = blockIdx.x * 256 + threadIdx.x;        // < 768*2048
    const int k = gid & 2047, f = gid >> 11;
    p5T[gid] = f2bf(p5[(size_t)k * 768 + f]);
}

// ---------------------------------------------------------------------------
// t2T[n][p][f] = bf16(x[n, f/3, (p>>6) + f%3 - 1, p&63])  (zero-padded rows)
// K-major unfold for the final bmm; scattered reads hit L2 (x is 67MB<192MB).
// ---------------------------------------------------------------------------
__global__ void build_t2T_kernel(const float* __restrict__ x,
                                 unsigned short* __restrict__ t2T) {
    const size_t gid = (size_t)blockIdx.x * 256 + threadIdx.x;  // < 16*4096*768
    const int f  = (int)(gid % 768);
    const size_t np = gid / 768;
    const int p  = (int)(np & 4095);
    const int n  = (int)(np >> 12);
    const int c2 = f / 3;
    const int k  = f - 3 * c2;
    const int h  = (p >> 6) + k - 1;
    const int j  = p & 63;
    float v = 0.f;
    if ((unsigned)h < 64u)
        v = x[(((size_t)n * 256 + c2) * 64 + h) * 64 + j];
    t2T[gid] = f2bf(v);
}

// ---------------------------------------------------------------------------
// t1[n,c,e,w] = sum_h x[n,c,h,w] * p1[h,e]
// ---------------------------------------------------------------------------
__global__ void t1_kernel(const float* __restrict__ x,
                          const float* __restrict__ p1,
                          float* __restrict__ t1) {
    const int slice = blockIdx.x;                // n*256 + c
    __shared__ float xs[64 * 64];
    __shared__ float ps[64 * 32];
    const int tid = threadIdx.x;
    const float* src = x + (size_t)slice * 4096;
#pragma unroll
    for (int j = 0; j < 16; ++j) xs[tid + j * 256] = src[tid + j * 256];
#pragma unroll
    for (int j = 0; j < 8; ++j) ps[tid + j * 256] = p1[tid + j * 256];
    __syncthreads();
    float* dst = t1 + (size_t)slice * 2048;
#pragma unroll
    for (int j = 0; j < 8; ++j) {
        const int i = tid + j * 256;
        const int e = i >> 6, w = i & 63;
        float s = 0.f;
#pragma unroll 8
        for (int h = 0; h < 64; ++h) s = fmaf(xs[h * 64 + w], ps[h * 32 + e], s);
        dst[i] = s;
    }
}

// ---------------------------------------------------------------------------
// t3: 9x9 depthwise conv over (e,w), pad 4, groups = 256
// ---------------------------------------------------------------------------
__global__ void t3_kernel(const float* __restrict__ t1,
                          const float* __restrict__ p3,
                          float* __restrict__ t3) {
    const int slice = blockIdx.x;                // n*256 + c
    const int c = slice & 255;
    __shared__ float sh[40 * 72];                // (32+8) x (64+8) halo tile
    __shared__ float w9[81];
    const int tid = threadIdx.x;
    for (int i = tid; i < 40 * 72; i += 256) sh[i] = 0.f;
    if (tid < 81) w9[tid] = p3[c * 81 + tid];
    __syncthreads();
    const float* src = t1 + (size_t)slice * 2048;
#pragma unroll
    for (int j = 0; j < 8; ++j) {
        const int i = tid + j * 256;
        const int e = i >> 6, w = i & 63;
        sh[(e + 4) * 72 + (w + 4)] = src[i];
    }
    __syncthreads();
    float* dst = t3 + (size_t)slice * 2048;
#pragma unroll
    for (int j = 0; j < 8; ++j) {
        const int i = tid + j * 256;
        const int e = i >> 6, w = i & 63;
        float s = 0.f;
#pragma unroll
        for (int de = 0; de < 9; ++de)
#pragma unroll
            for (int dw = 0; dw < 9; ++dw)
                s = fmaf(sh[(e + de) * 72 + (w + dw)], w9[de * 9 + dw], s);
        dst[i] = s;
    }
}

// ---------------------------------------------------------------------------
// t4: (5,1) conv, groups = 64 (4 in / 4 out channels per group), pad (2,0)
// ---------------------------------------------------------------------------
__global__ void t4_kernel(const float* __restrict__ t3,
                          const float* __restrict__ p4,
                          unsigned short* __restrict__ t4b) {
    const int bid = blockIdx.x;                  // n*64 + g
    const int n = bid >> 6, g = bid & 63;
    __shared__ float sh[4 * 36 * 64];            // 4 ch, e-halo of 2
    __shared__ float wg[80];
    const int tid = threadIdx.x;
#pragma unroll
    for (int j = 0; j < 36; ++j) sh[tid + j * 256] = 0.f;
    if (tid < 80) wg[tid] = p4[(size_t)(g * 4) * 20 + tid];
    __syncthreads();
    const float* src = t3 + ((size_t)n * 256 + g * 4) * 2048;
#pragma unroll
    for (int j = 0; j < 32; ++j) {
        const int i = tid + j * 256;
        const int ci = i >> 11, rem = i & 2047;
        const int e = rem >> 6, w = rem & 63;
        sh[ci * (36 * 64) + (e + 2) * 64 + w] = src[ci * 2048 + rem];
    }
    __syncthreads();
    unsigned short* dst = t4b + ((size_t)n * 256 + g * 4) * 2048;
#pragma unroll
    for (int j = 0; j < 32; ++j) {
        const int i = tid + j * 256;
        const int oc = i >> 11, rem = i & 2047;
        const int e = rem >> 6, w = rem & 63;
        float s = 0.f;
#pragma unroll
        for (int ci = 0; ci < 4; ++ci)
#pragma unroll
            for (int ke = 0; ke < 5; ++ke)
                s = fmaf(sh[ci * (36 * 64) + (e + ke) * 64 + w],
                         wg[oc * 20 + ci * 5 + ke], s);
        dst[oc * 2048 + rem] = f2bf(s);
    }
}

// ---------------------------------------------------------------------------
// Shared WMMA GEMM body: C(M x Ncols) = A(M x K) * B(K x Ncols)
// block tile 128x128, 8 waves as 4(M) x 2(N), wave tile 32x64:
// 2 A-frags x 4 B-frags -> 8 WMMA per K step.
// Double-buffered LDS fed by TDM (wave 0 issues; TENSORcnt + barrier).
// ---------------------------------------------------------------------------
template <int KTOT>
__device__ __forceinline__ void gemm_128x128(const unsigned short* Abase,
                                             int ldA,
                                             const unsigned short* BTbase,
                                             size_t ldk,
                                             float* Cbase, int ldc,
                                             unsigned short (*As)[128 * 32],
                                             unsigned short (*Bt)[128 * 32],
                                             int tid) {
    const int lane = tid & 31, wave = tid >> 5;
    const int wm = (wave & 3) * 32;
    const int wn = (wave >> 2) * 64;
    const v8f vzero = {0.f, 0.f, 0.f, 0.f, 0.f, 0.f, 0.f, 0.f};
    v8f acc[2][4];
#pragma unroll
    for (int mi = 0; mi < 2; ++mi)
#pragma unroll
        for (int ni = 0; ni < 4; ++ni) acc[mi][ni] = vzero;

    if (wave == 0) {
        tdm_load_tile(As[0], Abase, (unsigned)ldA);
        tdm_load_tile(Bt[0], BTbase, (unsigned)ldk);
    }
    int cur = 0;
    constexpr int NIT = KTOT / 32;
    for (int it = 0; it < NIT; ++it) {
        if (wave == 0) wait_tensor0();           // wave 0's DMAs are in LDS
        __syncthreads();                         // publish to all waves
        if (it + 1 < NIT && wave == 0) {         // kick next tile's DMAs
            const int k1 = (it + 1) * 32;
            tdm_load_tile(As[cur ^ 1], Abase + k1, (unsigned)ldA);
            tdm_load_tile(Bt[cur ^ 1], BTbase + k1, (unsigned)ldk);
        }
        const unsigned short* Ac = As[cur];
        const unsigned short* Bc = Bt[cur];
        v16bf a0 = load_frag_a(Ac, wm,      lane);
        v16bf a1 = load_frag_a(Ac, wm + 16, lane);
        v16bf b0 = load_frag_b(Bc, wn,      lane);
        v16bf b1 = load_frag_b(Bc, wn + 16, lane);
        v16bf b2 = load_frag_b(Bc, wn + 32, lane);
        v16bf b3 = load_frag_b(Bc, wn + 48, lane);
        acc[0][0] = wmma_bf16(a0, b0, acc[0][0]);
        acc[0][1] = wmma_bf16(a0, b1, acc[0][1]);
        acc[0][2] = wmma_bf16(a0, b2, acc[0][2]);
        acc[0][3] = wmma_bf16(a0, b3, acc[0][3]);
        acc[1][0] = wmma_bf16(a1, b0, acc[1][0]);
        acc[1][1] = wmma_bf16(a1, b1, acc[1][1]);
        acc[1][2] = wmma_bf16(a1, b2, acc[1][2]);
        acc[1][3] = wmma_bf16(a1, b3, acc[1][3]);
        cur ^= 1;
    }
    const int hi = (lane >> 4) * 8;
    const int cl = lane & 15;
#pragma unroll
    for (int mi = 0; mi < 2; ++mi)
#pragma unroll
        for (int ni = 0; ni < 4; ++ni) {
            v8f cc = acc[mi][ni];
            const size_t gr = wm + mi * 16 + hi;
            const int gc = wn + ni * 16 + cl;
#pragma unroll
            for (int v = 0; v < 8; ++v)
                Cbase[(gr + v) * (size_t)ldc + gc] = cc[v];
        }
}

// t5 GEMM: per n, C(256x768) = t4(256x2048) * p5(2048x768)
__global__ void __launch_bounds__(256)
t5_gemm_kernel(const unsigned short* __restrict__ A,
               const unsigned short* __restrict__ BT,   // p5T (768 x 2048)
               float* __restrict__ C) {
    const int bid = blockIdx.x;                  // 16 * 2 * 6 = 192
    const int n  = bid / 12;
    const int r  = bid % 12;
    const int m0 = (r / 6) * 128;
    const int n0 = (r % 6) * 128;
    __shared__ __align__(64) unsigned short As[2][128 * 32];
    __shared__ __align__(64) unsigned short Bt[2][128 * 32];
    gemm_128x128<2048>(A + ((size_t)n * 256 + m0) * 2048, 2048,
                       BT + (size_t)n0 * 2048, 2048,
                       C + ((size_t)n * 256 + m0) * 768 + n0, 768,
                       As, Bt, threadIdx.x);
}

// out GEMM: per n, C(256x4096) = t6(256x768) * t2(768x4096), B from t2T
__global__ void __launch_bounds__(256)
out_gemm_kernel(const unsigned short* __restrict__ A,
                const unsigned short* __restrict__ t2T,  // (16,4096,768)
                float* __restrict__ C) {
    const int bid = blockIdx.x;                  // 16 * 2 * 32 = 1024
    const int n  = bid / 64;
    const int r  = bid % 64;
    const int m0 = (r / 32) * 128;
    const int p0 = (r % 32) * 128;
    __shared__ __align__(64) unsigned short As[2][128 * 32];
    __shared__ __align__(64) unsigned short Bt[2][128 * 32];
    gemm_128x128<768>(A + ((size_t)n * 256 + m0) * 768, 768,
                      t2T + ((size_t)n * 4096 + p0) * 768, 768,
                      C + ((size_t)n * 256 + m0) * 4096 + p0, 4096,
                      As, Bt, threadIdx.x);
}

// ---------------------------------------------------------------------------
// t6: depthwise (1,3) over f, pad 1, folded output scale 1/sqrt(768) -> bf16
// ---------------------------------------------------------------------------
__global__ void t6_kernel(const float* __restrict__ t5,
                          const float* __restrict__ p6,
                          unsigned short* __restrict__ t6b) {
    const int gid = blockIdx.x * 256 + threadIdx.x;       // < 16*256*768
    if (gid >= 16 * 256 * 768) return;
    const int f = gid % 768;
    const int c = (gid / 768) & 255;
    const float* row = t5 + (gid - f);
    float s = p6[c * 3 + 1] * row[f];
    if (f > 0)   s = fmaf(p6[c * 3 + 0], row[f - 1], s);
    if (f < 767) s = fmaf(p6[c * 3 + 2], row[f + 1], s);
    t6b[gid] = f2bf(s * 0.03608439182435161f);            // 1/sqrt(768)
}

// ---------------------------------------------------------------------------
// host launcher
// ---------------------------------------------------------------------------
extern "C" void kernel_launch(void* const* d_in, const int* in_sizes, int n_in,
                              void* d_out, int out_size, void* d_ws,
                              size_t ws_size, hipStream_t stream) {
    (void)in_sizes; (void)n_in; (void)out_size; (void)ws_size;
    const float* x  = (const float*)d_in[0];   // (16,256,64,64)
    const float* p1 = (const float*)d_in[1];   // (64,32)
    const float* p3 = (const float*)d_in[2];   // (256,1,9,9)
    const float* p4 = (const float*)d_in[3];   // (256,4,5,1)
    const float* p5 = (const float*)d_in[4];   // (32,64,1,768)
    const float* p6 = (const float*)d_in[5];   // (256,1,1,3)
    float* out = (float*)d_out;                // (16,256,64,64)

    char* ws = (char*)d_ws;
    size_t off = 0;
    auto take = [&](size_t bytes) -> char* {
        char* p = ws + off;
        off += (bytes + 255) & ~(size_t)255;
        return p;
    };
    float*          t1  = (float*)take(8388608ull * 4);           // (16,256,32,64)
    float*          t3  = (float*)take(8388608ull * 4);           // (16,256,32,64)
    unsigned short* t4b = (unsigned short*)take(8388608ull * 2);  // bf16
    unsigned short* p5T = (unsigned short*)take(1572864ull * 2);  // (768,2048) bf16
    float*          t5  = (float*)take(3145728ull * 4);           // (16,256,768)
    unsigned short* t6b = (unsigned short*)take(3145728ull * 2);  // bf16
    unsigned short* t2T = (unsigned short*)take(50331648ull * 2); // (16,4096,768)

    build_p5T_kernel<<<6144, 256, 0, stream>>>(p5, p5T);
    build_t2T_kernel<<<196608, 256, 0, stream>>>(x, t2T);
    t1_kernel<<<4096, 256, 0, stream>>>(x, p1, t1);
    t3_kernel<<<4096, 256, 0, stream>>>(t1, p3, t3);
    t4_kernel<<<1024, 256, 0, stream>>>(t3, p4, t4b);
    t5_gemm_kernel<<<192, 256, 0, stream>>>(t4b, p5T, t5);
    t6_kernel<<<12288, 256, 0, stream>>>(t5, p6, t6b);
    out_gemm_kernel<<<1024, 256, 0, stream>>>(t6b, t2T, out);
}
